// CatanGNN_34815004901971
// MI455X (gfx1250) — matrix operand order repaired and checked by reading, not compiled, not run on hardware
//
#include <hip/hip_runtime.h>

typedef __attribute__((ext_vector_type(16))) _Float16 v16h;
typedef __attribute__((ext_vector_type(8)))  float    v8f;

#define HID 128

// type ids: 0=tile, 1=vertex, 2=edge
__constant__ int d_N[3]    = {19, 54, 72};
__constant__ int d_PAD[3]  = {32, 64, 80};   // rows padded to multiple of 16
__constant__ int d_XOFF[3] = {0, 32, 96};    // row offsets in padded f16 feature buffer (176 rows)
__constant__ int d_HOFF[3] = {0, 19, 73};    // row offsets in exact-row accumulator (145 rows)
__constant__ int d_SOFF[3] = {0, 304, 1168}; // float offsets in async staging buffer
// relations in sorted-key order: E2E,E2T,E2V,T2E,T2T,T2V,V2E,V2T,V2V
__constant__ int d_ST[9] = {2,2,2,0,0,0,1,1,1};
__constant__ int d_DT[9] = {2,0,1,2,0,1,2,0,1};
__constant__ int d_EN[9] = {288,114,144,114,84,114,144,114,144};
__constant__ int d_SL[9] = {1,0,0,0,1,0,0,0,1};

struct RelP { const float *Wl, *Wr, *a, *bias, *bl, *br; };

struct Params {
  const float* x[3];        // x_tile, x_vertex, x_edge  [B, N_t, 16]
  const int*   e[9];        // edge index [2, E] per relation (sorted rel order)
  RelP c1[9], c2[9];
  const float* id_emb[3];   // [N_t, 16]
  const float* projW[3];    // [128, 256]
  const float* projb[3];    // [256]
  const float *ro1W, *ro1b; // [768,256],[256]
  const float *ro2W, *ro2b; // [256,256],[256]
  float* out;               // [B, 256]
};

// ---- CDNA5 async global->LDS copy (ASYNCcnt-tracked, no VGPR staging) ----
__device__ __forceinline__ void async_copy_b32(unsigned lds_off, const float* gaddr) {
  asm volatile("global_load_async_to_lds_b32 %0, %1, off"
               :: "v"(lds_off), "v"(gaddr)
               : "memory");
}
__device__ __forceinline__ void wait_asynccnt0() {
  asm volatile("s_wait_asynccnt 0x0" ::: "memory");
}

// ---- WMMA fragment loaders (16-bit 16x16x32, layouts per CDNA5 ISA 7.12.2) ----

// A: 16x32 f16 tile from LDS, row-major with stride HID halves.
__device__ __forceinline__ v16h load_A_frag(const _Float16* X, int row0, int kbase, int lane) {
  const _Float16* base = X + (size_t)(row0 + (lane & 15)) * HID;
  int hi8 = (lane >> 4) << 3;     // lanes 16-31 hold K+8 (and K+24) halves of each row
  v16h a;
#pragma unroll
  for (int v = 0; v < 8; ++v) {
    int k = kbase + ((v & 4) << 2) + hi8 + ((v & 3) << 1);
    a[2 * v]     = base[k];
    a[2 * v + 1] = base[k + 1];
  }
  return a;
}

// B: 32x16 tile from global f32 weights W[K][128] (row-major), converted to f16.
// lanes 0-15: col=lane, K=kbase+0..7,16..23 ; lanes 16-31: col=lane-16, K=kbase+8..15,24..31
__device__ __forceinline__ v16h load_B_frag(const float* W, int col0, int kbase, int lane) {
  int col = col0 + (lane & 15);
  int k0  = kbase + ((lane >> 4) << 3);
  v16h b;
#pragma unroll
  for (int v = 0; v < 8; ++v) {
    int k = k0 + ((v & 4) << 2) + ((v & 3) << 1);
    b[2 * v]     = (_Float16)W[(size_t)k * HID + col];
    b[2 * v + 1] = (_Float16)W[(size_t)(k + 1) * HID + col];
  }
  return b;
}

// Out[Mpad x 128] (f16, LDS) = X[xrow0.., 0..K) @ W[K][128] + bias.  8 waves cooperate.
__device__ __forceinline__ void gemm_f16(const _Float16* Xlds, int xrow0,
                                         const float* W, const float* bias, int K,
                                         _Float16* Out, int Mpad, int tid) {
  int wave = tid >> 5, lane = tid & 31;
  int tiles = (Mpad >> 4) << 3;  // Mtiles * 8 ntiles; always a multiple of 8
  for (int t = wave; t < tiles; t += 8) {
    int mt = t >> 3, nt = t & 7;
    int col = (nt << 4) + (lane & 15);
    float bv = bias[col];
    v8f acc = {bv, bv, bv, bv, bv, bv, bv, bv};
    for (int kb = 0; kb < K; kb += 32) {
      v16h a = load_A_frag(Xlds, xrow0 + (mt << 4), kb, lane);
      v16h b = load_B_frag(W, nt << 4, kb, lane);
      acc = __builtin_amdgcn_wmma_f32_16x16x32_f16(false, a, false, b, (short)0, acc,
                                                   false, false);
    }
    int hi = lane >> 4;
#pragma unroll
    for (int v = 0; v < 8; ++v) {
      int row = (mt << 4) + v + (hi << 3);  // C/D layout: VGPR v -> M = v (+8 for hi lanes)
      Out[(size_t)row * HID + col] = (_Float16)acc[v];
    }
  }
}

__global__ __launch_bounds__(256, 1) void catan_gnn_kernel(Params p) {
  const int b = blockIdx.x;
  const int tid = threadIdx.x;

  __shared__ _Float16 sX[176 * HID];  // current-layer node features (f16, padded rows)
  __shared__ float    sH[145 * HID];  // hetero-conv f32 accumulator (exact rows)
  __shared__ union {
    struct {
      _Float16 XL[80 * HID];  // Wl-transformed src features (per relation)
      _Float16 XR[80 * HID];  // Wr-transformed dst features
      float esc[360];         // per-edge attention logits
      int   eidx[360];        // packed src | dst<<16
      float m[80];            // per-dst softmax max
      float den[80];          // per-dst softmax denom
    } c;
    struct {
      float hmean[3 * HID];
      float pooled[768];
      float z[256];
    } h;
    struct {
      float stage[145 * 16];  // raw f32 inputs landed via async-to-LDS
    } s;
  } u;

  // ---- phase 0a: async DMA of this graph's raw features into LDS (ASYNCcnt) ----
  for (int t = 0; t < 3; ++t) {
    const float* src = p.x[t] + (size_t)b * d_N[t] * 16;
    const int n = d_N[t] * 16;
    const int so = d_SOFF[t];
    for (int i = tid; i < n; i += 256) {
      unsigned lds_off = (unsigned)(uintptr_t)(&u.s.stage[so + i]);
      async_copy_b32(lds_off, src + i);
    }
  }
  wait_asynccnt0();
  __syncthreads();

  // ---- phase 0b: build layer-1 input  [x | id_emb] -> f16, zero padded rows ----
  for (int t = 0; t < 3; ++t) {
    const int Nt = d_N[t];
    const float* xb = &u.s.stage[d_SOFF[t]];
    for (int i = tid; i < d_PAD[t] * 32; i += 256) {
      int r = i >> 5, c = i & 31;
      _Float16 v = (_Float16)0.0f;
      if (r < Nt) v = (_Float16)(c < 16 ? xb[r * 16 + c] : p.id_emb[t][r * 16 + (c - 16)]);
      sX[(size_t)(d_XOFF[t] + r) * HID + c] = v;
    }
  }
  __syncthreads();

  // ---- two GATv2 hetero-conv layers ----
  for (int L = 0; L < 2; ++L) {
    const int K = L ? HID : 32;
    for (int i = tid; i < 145 * HID; i += 256) sH[i] = 0.0f;
    __syncthreads();

    for (int r = 0; r < 9; ++r) {
      const RelP rp = L ? p.c2[r] : p.c1[r];
      const int st = d_ST[r], dt = d_DT[r], E = d_EN[r];
      const int Nd = d_N[dt];
      const int Etot = E + (d_SL[r] ? Nd : 0);

      // prefetch next relation's weights into cache (global_prefetch_b8)
      if (r < 8) {
        const RelP rn = L ? p.c2[r + 1] : p.c1[r + 1];
        __builtin_prefetch(rn.Wl + (tid << 4), 0, 0);
        __builtin_prefetch(rn.Wr + (tid << 4), 0, 0);
      }

      gemm_f16(sX, d_XOFF[st], rp.Wl, rp.bl, K, u.c.XL, d_PAD[st], tid);
      gemm_f16(sX, d_XOFF[dt], rp.Wr, rp.br, K, u.c.XR, d_PAD[dt], tid);
      __syncthreads();

      // attention logits: e = a . leakyrelu(xl[s] + xr[d], 0.2)
      for (int e = tid; e < Etot; e += 256) {
        int s, d;
        if (e < E) { s = p.e[r][e]; d = p.e[r][E + e]; }
        else       { s = d = e - E; }
        u.c.eidx[e] = s | (d << 16);
        float acc = 0.0f;
        for (int c = 0; c < HID; ++c) {
          float v = (float)u.c.XL[(size_t)s * HID + c] + (float)u.c.XR[(size_t)d * HID + c];
          v = v > 0.0f ? v : 0.2f * v;
          acc += rp.a[c] * v;
        }
        u.c.esc[e] = acc;
      }
      __syncthreads();

      // segment softmax stats per destination node
      for (int n = tid; n < Nd; n += 256) {
        float m = -1e30f;
        for (int e = 0; e < Etot; ++e)
          if ((u.c.eidx[e] >> 16) == n) m = fmaxf(m, u.c.esc[e]);
        if (m < -0.9e30f) m = 0.0f;  // empty segment (matches isfinite guard)
        float den = 0.0f;
        for (int e = 0; e < Etot; ++e)
          if ((u.c.eidx[e] >> 16) == n) den += __expf(u.c.esc[e] - m);
        u.c.m[n]   = m;
        u.c.den[n] = den > 0.0f ? den : 1.0f;
      }
      __syncthreads();

      // aggregate: H[dt][d] += alpha * xl[s]   (LDS ds_add_f32 atomics)
      for (int i = tid; i < Etot * 4; i += 256) {
        int e = i >> 2, cb = (i & 3) << 5;
        int pk = u.c.eidx[e];
        int s = pk & 0xffff, d = pk >> 16;
        float alpha = __expf(u.c.esc[e] - u.c.m[d]) / u.c.den[d];
        float* Hd = &sH[(size_t)(d_HOFF[dt] + d) * HID + cb];
        const _Float16* xl = &u.c.XL[(size_t)s * HID + cb];
        for (int c = 0; c < 32; ++c) atomicAdd(&Hd[c], alpha * (float)xl[c]);
      }
      __syncthreads();

      // per-relation output bias, added once per dst node
      for (int i = tid; i < Nd * HID; i += 256)
        sH[(size_t)d_HOFF[dt] * HID + i] += rp.bias[i & (HID - 1)];
      __syncthreads();
    }

    if (L == 0) {
      // ELU, convert to f16 next-layer features (zero the padded rows)
      for (int t = 0; t < 3; ++t) {
        for (int i = tid; i < d_PAD[t] * HID; i += 256) {
          int r = i >> 7, c = i & (HID - 1);
          _Float16 v = (_Float16)0.0f;
          if (r < d_N[t]) {
            float h = sH[(size_t)(d_HOFF[t] + r) * HID + c];
            v = (_Float16)(h > 0.0f ? h : (__expf(h) - 1.0f));
          }
          sX[(size_t)(d_XOFF[t] + r) * HID + c] = v;
        }
      }
      __syncthreads();
    } else {
      // final ELU in place (f32, feeds pooling)
      for (int i = tid; i < 145 * HID; i += 256) {
        float h = sH[i];
        sH[i] = h > 0.0f ? h : (__expf(h) - 1.0f);
      }
      __syncthreads();
    }
  }

  // ---- head: mean-pool, project, 2-layer MLP (mean commutes with linear map) ----
  for (int i = tid; i < 3 * HID; i += 256) {
    int t = i >> 7, c = i & (HID - 1);
    float s = 0.0f;
    for (int r = 0; r < d_N[t]; ++r) s += sH[(size_t)(d_HOFF[t] + r) * HID + c];
    u.h.hmean[i] = s / (float)d_N[t];
  }
  __syncthreads();

  for (int i = tid; i < 768; i += 256) {     // pooled order: tile, vertex, edge == type ids
    int t = i >> 8, c = i & 255;
    const float* W = p.projW[t];
    float s = p.projb[t][c];
    for (int k = 0; k < HID; ++k) s += u.h.hmean[t * HID + k] * W[(size_t)k * 256 + c];
    u.h.pooled[i] = s;
  }
  __syncthreads();

  for (int c = tid; c < 256; c += 256) {
    float s = p.ro1b[c];
    for (int k = 0; k < 768; ++k) s += u.h.pooled[k] * p.ro1W[(size_t)k * 256 + c];
    u.h.z[c] = fmaxf(s, 0.0f);
  }
  __syncthreads();

  for (int c = tid; c < 256; c += 256) {
    float s = p.ro2b[c];
    for (int k = 0; k < 256; ++k) s += u.h.z[k] * p.ro2W[(size_t)k * 256 + c];
    p.out[(size_t)b * 256 + c] = s;
  }
}

extern "C" void kernel_launch(void* const* d_in, const int* in_sizes, int n_in,
                              void* d_out, int out_size, void* d_ws, size_t ws_size,
                              hipStream_t stream) {
  (void)n_in; (void)d_ws; (void)ws_size; (void)out_size;
  Params P;
  P.x[0] = (const float*)d_in[0];   // x_tile
  P.x[1] = (const float*)d_in[1];   // x_vertex
  P.x[2] = (const float*)d_in[2];   // x_edge
  // edge arrays, mapped to sorted rel order E2E,E2T,E2V,T2E,T2T,T2V,V2E,V2T,V2V
  const int emap[9] = {11, 9, 10, 5, 3, 4, 8, 6, 7};
  for (int r = 0; r < 9; ++r) P.e[r] = (const int*)d_in[emap[r]];
  // params pytree (sorted keys): conv1[rel]{Wl,Wr,a,bias,bl,br} ... conv2 ...
  for (int r = 0; r < 9; ++r) {
    int b1 = 12 + r * 6;
    P.c1[r] = {(const float*)d_in[b1 + 0], (const float*)d_in[b1 + 1],
               (const float*)d_in[b1 + 2], (const float*)d_in[b1 + 3],
               (const float*)d_in[b1 + 4], (const float*)d_in[b1 + 5]};
    int b2 = 66 + r * 6;
    P.c2[r] = {(const float*)d_in[b2 + 0], (const float*)d_in[b2 + 1],
               (const float*)d_in[b2 + 2], (const float*)d_in[b2 + 3],
               (const float*)d_in[b2 + 4], (const float*)d_in[b2 + 5]};
  }
  // id_emb sorted keys: edge(120), tile(121), vertex(122) -> type ids 0,1,2
  P.id_emb[0] = (const float*)d_in[121];
  P.id_emb[1] = (const float*)d_in[122];
  P.id_emb[2] = (const float*)d_in[120];
  // proj sorted keys: edge{W,b}(123,124), tile(125,126), vertex(127,128)
  P.projW[0] = (const float*)d_in[125]; P.projb[0] = (const float*)d_in[126];
  P.projW[1] = (const float*)d_in[127]; P.projb[1] = (const float*)d_in[128];
  P.projW[2] = (const float*)d_in[123]; P.projb[2] = (const float*)d_in[124];
  P.ro1W = (const float*)d_in[129]; P.ro1b = (const float*)d_in[130];
  P.ro2W = (const float*)d_in[131]; P.ro2b = (const float*)d_in[132];
  P.out = (float*)d_out;

  const int B = in_sizes[0] / (19 * 16);
  catan_gnn_kernel<<<B, 256, 0, stream>>>(P);
}
